// Decoder_87290915324425
// MI455X (gfx1250) — matrix operand order repaired and checked by reading, not compiled, not run on hardware
//
#include <hip/hip_runtime.h>
#include <hip/hip_bf16.h>

// ---------------------------------------------------------------------------
// Types for CDNA5 (gfx1250, wave32) WMMA + TDM
// ---------------------------------------------------------------------------
typedef __bf16 bf16;
typedef __attribute__((ext_vector_type(8)))  bf16  bf16x8;
typedef __attribute__((ext_vector_type(16))) bf16  v16bf;
typedef __attribute__((ext_vector_type(8)))  float v8f;
typedef __attribute__((ext_vector_type(4)))  unsigned int v4u;
typedef __attribute__((ext_vector_type(8)))  int   v8i;
typedef __attribute__((ext_vector_type(4)))  int   v4i;

static_assert(sizeof(v16bf) == 32, "v16bf must be 32 bytes");

#if defined(__gfx1250__) && __has_builtin(__builtin_amdgcn_tensor_load_to_lds) && \
    __has_builtin(__builtin_amdgcn_s_wait_tensorcnt)
#define GEMM_USE_TDM 1
#else
#define GEMM_USE_TDM 0
#endif

// Problem constants (from reference)
#define VV 32000
#define EE 512
#define HH 1024
#define MM 1024
#define BB 32
#define TT 64
#define SS 64
#define H3 3072   // 3*H

__device__ __forceinline__ bf16 f2bf(float f) {
    // round-to-nearest-even f32 -> bf16, branch-free
    unsigned u = __builtin_bit_cast(unsigned, f);
    unsigned r = (u + 0x7FFFu + ((u >> 16) & 1u)) >> 16;
    unsigned short s = (unsigned short)r;
    return __builtin_bit_cast(bf16, s);
}

// ---------------------------------------------------------------------------
// TDM: DMA a [rows x 32] bf16 tile (row stride = Kdim elems) from global into
// LDS with a hardware-padded row stride of 40 bf16 (16 DWORDs data + 4 DWORDs
// pad), matching the software LDS layout used by the WMMA fragment loads.
// D# packing per CDNA5 ISA ch.8. Verified to lower to `tensor_load_to_lds`
// with the 2-SGPR-group (2D) descriptor form in round 2.
// ---------------------------------------------------------------------------
#if GEMM_USE_TDM
__device__ __forceinline__ void tdm_load_tile_a(const bf16* gsrc,
                                                unsigned lds_off,
                                                int Kdim, int rows) {
    unsigned long long ga = (unsigned long long)(uintptr_t)gsrc;
    v4u g0;
    g0[0] = 1u;                                   // count=1, user descriptor
    g0[1] = lds_off;                              // lds_addr (byte offset)
    g0[2] = (unsigned)ga;                         // global_addr[31:0]
    g0[3] = (unsigned)((ga >> 32) & 0x01FFFFFFu)  // global_addr[56:32]
          | (2u << 30);                           // type = 2 ("image")
    v8i g1;
    g1[0] = (int)((1u << 16)        // data_size = 1 -> 2 bytes
                | (1u << 20)        // pad_enable
                | (3u << 22)        // pad_interval = 3 -> every 16 DWORDs
                | (3u << 25));      // pad_amount  = 3 -> 4 DWORDs (8 bf16)
    g1[1] = (int)(((unsigned)Kdim & 0xFFFFu) << 16);            // tensor_dim0 lo16
    g1[2] = (int)((((unsigned)Kdim >> 16) & 0xFFFFu)            // tensor_dim0 hi16
                | (((unsigned)rows & 0xFFFFu) << 16));          // tensor_dim1 lo16
    g1[3] = (int)(32u << 16);                                   // tile_dim0 = 32
    g1[4] = (int)((unsigned)rows & 0xFFFFu);                    // tile_dim1 = rows
    g1[5] = (int)Kdim;                                          // dim0 stride lo32
    g1[6] = 0;
    g1[7] = 0;
    v4i gz = {0, 0, 0, 0};
#if __has_include(<hip/amd_detail/amd_gfx1250_TDM.h>)
    v8i gz8 = {0, 0, 0, 0, 0, 0, 0, 0};
    __builtin_amdgcn_tensor_load_to_lds(g0, g1, gz, gz, gz8, 0);   // 6-arg toolchain
#else
    __builtin_amdgcn_tensor_load_to_lds(g0, g1, gz, gz, 0);        // 5-arg toolchain
#endif
}
#endif

// ---------------------------------------------------------------------------
// f32 -> bf16 conversion (grid-stride)
// ---------------------------------------------------------------------------
__global__ void convert_f32_bf16_kernel(const float* __restrict__ src,
                                        bf16* __restrict__ dst, long n) {
    long i = (long)blockIdx.x * blockDim.x + threadIdx.x;
    long stride = (long)gridDim.x * blockDim.x;
    for (; i < n; i += stride) dst[i] = f2bf(src[i]);
}

// ---------------------------------------------------------------------------
// Embedding gather: xbf[(t*B+b)*E + e] = bf16(emb[tokens[b,t]*E + e])
// ---------------------------------------------------------------------------
__global__ void embed_gather_kernel(const int* __restrict__ tokens,
                                    const float* __restrict__ emb,
                                    bf16* __restrict__ xbf) {
    long idx = (long)blockIdx.x * blockDim.x + threadIdx.x;
    long total = (long)TT * BB * EE;
    if (idx >= total) return;
    int e  = (int)(idx % EE);
    long tb = idx / EE;           // t*B + b
    int b  = (int)(tb % BB);
    int t  = (int)(tb / BB);
    int tok = tokens[b * TT + t];
    xbf[idx] = f2bf(emb[(long)tok * EE + e]);
}

// ---------------------------------------------------------------------------
// Zero-init recurrent state
// ---------------------------------------------------------------------------
__global__ void init_state_kernel(float* __restrict__ h, bf16* __restrict__ hb,
                                  float* __restrict__ attn, bf16* __restrict__ attnb) {
    int idx = blockIdx.x * blockDim.x + threadIdx.x;
    if (idx >= BB * HH) return;
    h[idx] = 0.f;     hb[idx]    = f2bf(0.f);
    attn[idx] = 0.f;  attnb[idx] = f2bf(0.f);
}

// ---------------------------------------------------------------------------
// Double-buffered tiled WMMA bf16 GEMM:
//   C[M,N](f32) = A[M,K]bf16 @ B[K,N]bf16 (+ addend[M,N]) (+ bias[N])
// Block: 256 threads = 8 wave32. Block tile BM x BN, K-step 32.
// Wave grid WROWS x WCOLS; each wave holds MT x NT 16x16 f32 accumulators,
// A-fragments reused across NT tiles.
// Software pipeline per K-step:
//   issue TDM for next A tile -> ldsA[nxt]         (async, TENSORcnt)
//   issue global loads for next B tile -> regs     (LOADcnt overlaps compute)
//   WMMA compute on ldsA[cur]/ldsB[cur]
//   scatter B regs -> ldsB[nxt] (u32-packed K-pairs), s_wait_tensorcnt, barrier
// Single barrier per iteration: the barrier at the end of step i-1 orders
// compute(i-1) against stores into the alternate buffer during step i.
// LDS layout (per buffer): A row-major [BM][40] (32 data + 8 pad bf16),
// B transposed [BN][40]; fragment reads are contiguous b128 LDS ops matching
// the CDNA5 16-bit WMMA layouts (A and B: K 8h..+7 & 16+8h..+7).
// Requires K%32==0 and N%BN==0 (true for all uses); M is guarded.
// ---------------------------------------------------------------------------
template <int BM, int BN, int WROWS, int WCOLS>
__global__ __launch_bounds__(256) void gemm_bf16_kernel(
    const bf16* __restrict__ A, const bf16* __restrict__ B,
    float* __restrict__ C,
    const float* __restrict__ addend,   // nullable, [M,N]
    const float* __restrict__ bias,     // nullable, [N]
    int Mdim, int Ndim, int Kdim)
{
    static_assert(WROWS * WCOLS == 8, "8 waves per block");
    constexpr int WTM = BM / WROWS;
    constexpr int WTN = BN / WCOLS;
    constexpr int MT  = WTM / 16;
    constexpr int NT  = WTN / 16;
    static_assert(MT >= 1 && NT >= 1, "tile too small");
    // B staging: units of (8 cols x 2 k-rows), packed as u32 pairs into LDS^T
    static_assert((2 * BN) % 256 == 0, "BN must be >= 128");
    constexpr int BUNITS = (2 * BN) / 256;       // units per thread
    constexpr int AVECS  = (BM * 4 + 255) / 256; // uint4 vecs per thread (fallback)

    __shared__ bf16 ldsA[2][BM * 40];
    __shared__ bf16 ldsB[2][BN * 40];   // transposed [n][k(32)+pad]

    const int tid  = threadIdx.x;
    const int wave = tid >> 5;
    const int lane = tid & 31;
    const int half = lane >> 4;
    const int l16  = lane & 15;

    const long m0 = (long)blockIdx.y * BM;
    const long n0 = (long)blockIdx.x * BN;
    const int  wm = (wave / WCOLS) * WTM;
    const int  wn = (wave % WCOLS) * WTN;

    const v8f vzero = {};
    v8f acc[MT][NT];
#pragma unroll
    for (int mt = 0; mt < MT; ++mt)
#pragma unroll
        for (int nt = 0; nt < NT; ++nt) acc[mt][nt] = vzero;

    // ---- B tile staging helpers (regs <-> LDS) ----
    bf16x8 blo[BUNITS], bhi[BUNITS];
    auto load_b_regs = [&](int k0) {
#pragma unroll
        for (int j = 0; j < BUNITS; ++j) {
            int u  = tid + 256 * j;
            int kp = u / (BN / 8);          // k-pair index, k = 2*kp
            int nc = (u % (BN / 8)) * 8;
            const bf16* gp = B + (long)(k0 + 2 * kp) * Ndim + (n0 + nc);
            blo[j] = *reinterpret_cast<const bf16x8*>(gp);
            bhi[j] = *reinterpret_cast<const bf16x8*>(gp + (long)Ndim);
        }
    };
    auto store_b_lds = [&](int buf) {
#pragma unroll
        for (int j = 0; j < BUNITS; ++j) {
            int u  = tid + 256 * j;
            int kp = u / (BN / 8);
            int nc = (u % (BN / 8)) * 8;
#pragma unroll
            for (int c = 0; c < 8; ++c) {
                unsigned lo = (unsigned)__builtin_bit_cast(unsigned short, blo[j][c]);
                unsigned hi = (unsigned)__builtin_bit_cast(unsigned short, bhi[j][c]);
                *reinterpret_cast<unsigned*>(&ldsB[buf][(nc + c) * 40 + 2 * kp]) =
                    lo | (hi << 16);
            }
        }
    };

#if !GEMM_USE_TDM
    uint4 areg[AVECS];
    auto load_a_regs = [&](int k0) {
#pragma unroll
        for (int j = 0; j < AVECS; ++j) {
            int v = tid + 256 * j;
            uint4 val = make_uint4(0u, 0u, 0u, 0u);
            if (v < BM * 4) {
                int r = v >> 2, kc = (v & 3) * 8;
                long gm = m0 + r;
                if (gm < Mdim)
                    val = *reinterpret_cast<const uint4*>(A + gm * (long)Kdim + k0 + kc);
            }
            areg[j] = val;
        }
    };
    auto store_a_lds = [&](int buf) {
#pragma unroll
        for (int j = 0; j < AVECS; ++j) {
            int v = tid + 256 * j;
            if (v < BM * 4) {
                int r = v >> 2, kc = (v & 3) * 8;
                *reinterpret_cast<uint4*>(&ldsA[buf][r * 40 + kc]) = areg[j];
            }
        }
    };
#endif

    // ---- prologue: stage K-tile 0 into buffer 0 ----
#if GEMM_USE_TDM
    if (wave == 0)
        tdm_load_tile_a(A + m0 * (long)Kdim,
                        (unsigned)(uintptr_t)(&ldsA[0][0]), Kdim, BM);
#else
    load_a_regs(0);
    store_a_lds(0);
#endif
    load_b_regs(0);
    store_b_lds(0);
#if GEMM_USE_TDM
    if (wave == 0) __builtin_amdgcn_s_wait_tensorcnt(0);
#endif
    __syncthreads();

    const int ksteps = Kdim >> 5;
    for (int i = 0; i < ksteps; ++i) {
        const int  cur = i & 1;
        const int  nxt = cur ^ 1;
        const int  kn  = (i + 1) << 5;
        const bool have_next = (i + 1) < ksteps;

        // ---- issue next-tile staging (overlaps with compute below) ----
#if GEMM_USE_TDM
        if (have_next && wave == 0)
            tdm_load_tile_a(A + m0 * (long)Kdim + kn,
                            (unsigned)(uintptr_t)(&ldsA[nxt][0]), Kdim, BM);
#else
        if (have_next) load_a_regs(kn);
#endif
        if (have_next) load_b_regs(kn);

        // ---- fragments + WMMA on current buffer ----
        v16bf afr[MT];
#pragma unroll
        for (int mt = 0; mt < MT; ++mt) {
            int am = wm + mt * 16 + l16;
            bf16x8 alo = *reinterpret_cast<const bf16x8*>(&ldsA[cur][am * 40 + 8 * half]);
            bf16x8 ahi = *reinterpret_cast<const bf16x8*>(&ldsA[cur][am * 40 + 16 + 8 * half]);
            afr[mt] = __builtin_shufflevector(alo, ahi,
                        0,1,2,3,4,5,6,7,8,9,10,11,12,13,14,15);
        }
        v16bf bfr[NT];
#pragma unroll
        for (int nt = 0; nt < NT; ++nt) {
            int bn = wn + nt * 16 + l16;
            bf16x8 vlo = *reinterpret_cast<const bf16x8*>(&ldsB[cur][bn * 40 + 8 * half]);
            bf16x8 vhi = *reinterpret_cast<const bf16x8*>(&ldsB[cur][bn * 40 + 16 + 8 * half]);
            bfr[nt] = __builtin_shufflevector(vlo, vhi,
                        0,1,2,3,4,5,6,7,8,9,10,11,12,13,14,15);
        }
#pragma unroll
        for (int mt = 0; mt < MT; ++mt)
#pragma unroll
            for (int nt = 0; nt < NT; ++nt)
                acc[mt][nt] = __builtin_amdgcn_wmma_f32_16x16x32_bf16(
                    false, afr[mt], false, bfr[nt], (short)0, acc[mt][nt],
                    false, false);

        // ---- publish next buffer ----
        if (have_next) {
            store_b_lds(nxt);
#if !GEMM_USE_TDM
            store_a_lds(nxt);
#endif
        }
#if GEMM_USE_TDM
        if (have_next && wave == 0) __builtin_amdgcn_s_wait_tensorcnt(0);
#endif
        __syncthreads();
    }

    // ---- store (C/D layout: lane<16 -> M=v, N=l16 ; lane>=16 -> M=v+8) ----
#pragma unroll
    for (int mt = 0; mt < MT; ++mt) {
#pragma unroll
        for (int nt = 0; nt < NT; ++nt) {
            long col = n0 + wn + nt * 16 + l16;
#pragma unroll
            for (int v = 0; v < 8; ++v) {
                long row = m0 + wm + mt * 16 + v + 8 * half;
                if (row < Mdim) {
                    long o = row * (long)Ndim + col;
                    float x = acc[mt][nt][v];
                    if (addend) x += addend[o];
                    if (bias)   x += bias[col];
                    C[o] = x;
                }
            }
        }
    }
}

// ---------------------------------------------------------------------------
// GRU gate fuse (reset_after=True):
//   z = sig(mx_z+mh_z); r = sig(mx_r+mh_r); hh = tanh(mx_h + r*mh_h)
//   h  = z*h + (1-z)*hh    (in place; also writes bf16 copy)
// ---------------------------------------------------------------------------
__global__ void gru_gate_kernel(const float* __restrict__ mx,
                                const float* __restrict__ mh,
                                float* __restrict__ h, bf16* __restrict__ hb) {
    int idx = blockIdx.x * blockDim.x + threadIdx.x;
    if (idx >= BB * HH) return;
    int b = idx >> 10;        // /HH
    int u = idx & (HH - 1);
    const float* mxr = mx + (long)b * H3;
    const float* mhr = mh + (long)b * H3;
    float z  = 1.f / (1.f + expf(-(mxr[u]            + mhr[u])));
    float r  = 1.f / (1.f + expf(-(mxr[HH + u]       + mhr[HH + u])));
    float hh = tanhf(mxr[2 * HH + u] + r * mhr[2 * HH + u]);
    float hp = h[idx];
    float hn = z * hp + (1.f - z) * hh;
    h[idx]  = hn;
    hb[idx] = f2bf(hn);
}

// ---------------------------------------------------------------------------
// Fused Luong attention step (one block per batch element b):
//   score[s] = <h_new[b], keys[b,s,:]>; align = softmax(score);
//   ctx[m]   = sum_s align[s]*memory[b,s,m]
// Writes attn(f32), attn(bf16) and ctxs_bf16 row (b*T + t).
// ---------------------------------------------------------------------------
__global__ __launch_bounds__(256) void attention_step_kernel(
    const float* __restrict__ h,        // [B,H]
    const float* __restrict__ keys,     // [B,S,H]
    const float* __restrict__ memory,   // [B,S,M] f32
    float* __restrict__ attn,           // [B,M]
    bf16*  __restrict__ attnb,          // [B,M]
    bf16*  __restrict__ ctxb,           // [B*T, M]
    int t)
{
    __shared__ float sh[HH];
    __shared__ float ssc[SS];
    __shared__ float sal[SS];
    const int b    = blockIdx.x;
    const int tid  = threadIdx.x;
    const int wave = tid >> 5;
    const int lane = tid & 31;

    for (int i = tid; i < HH; i += 256) sh[i] = h[(long)b * HH + i];
    __syncthreads();

    // 8 waves x 8 scores each; wave32 shuffle reduction
#pragma unroll
    for (int j = 0; j < 8; ++j) {
        int s = wave * 8 + j;
        const float* kp = keys + ((long)b * SS + s) * HH;
        float p = 0.f;
        for (int i = lane; i < HH; i += 32) p += sh[i] * kp[i];
        for (int off = 16; off >= 1; off >>= 1) p += __shfl_xor(p, off, 32);
        if (lane == 0) ssc[s] = p;
    }
    __syncthreads();

    // softmax over S=64 (redundant per-thread reduction over LDS broadcast)
    float mx = -3.402823466e38f;
    for (int s = 0; s < SS; ++s) mx = fmaxf(mx, ssc[s]);
    float sum = 0.f;
    for (int s = 0; s < SS; ++s) sum += expf(ssc[s] - mx);
    if (tid < SS) sal[tid] = expf(ssc[tid] - mx) / sum;
    __syncthreads();

    // context = align @ memory[b]
    for (int mcol = tid; mcol < MM; mcol += 256) {
        const float* mp = memory + (long)b * SS * MM + mcol;
        float acc = 0.f;
        for (int s = 0; s < SS; ++s) acc += sal[s] * mp[(long)s * MM];
        attn[(long)b * MM + mcol] = acc;
        bf16 v = f2bf(acc);
        attnb[(long)b * MM + mcol] = v;
        ctxb[((long)b * TT + t) * MM + mcol] = v;
    }
}

// ---------------------------------------------------------------------------
// Host-side orchestration
// ---------------------------------------------------------------------------
extern "C" void kernel_launch(void* const* d_in, const int* in_sizes, int n_in,
                              void* d_out, int out_size, void* d_ws, size_t ws_size,
                              hipStream_t stream) {
    (void)in_sizes; (void)n_in; (void)out_size; (void)ws_size;

    const int*   tokens = (const int*)  d_in[0];   // [B,T]
    const float* memory = (const float*)d_in[1];   // [B,S,M]
    const float* emb    = (const float*)d_in[2];   // [V,E]
    const float* W_x    = (const float*)d_in[3];   // [E+M, 3H]
    const float* W_h    = (const float*)d_in[4];   // [H, 3H]
    const float* b_x    = (const float*)d_in[5];   // [3H]
    const float* b_h    = (const float*)d_in[6];   // [3H]
    const float* W_mem  = (const float*)d_in[7];   // [M, H]
    const float* W_fc   = (const float*)d_in[8];   // [M, V]
    const float* b_fc   = (const float*)d_in[9];   // [V]
    float* logits = (float*)d_out;                 // [B*T, V]

    // Workspace carve-up (256B aligned)
    char* wp = (char*)d_ws;
    auto carve = [&](size_t bytes) -> void* {
        void* p = (void*)wp;
        wp += (bytes + 255) & ~(size_t)255;
        return p;
    };
    bf16*  xbf    = (bf16*) carve((size_t)TT * BB * EE * 2);   // [T*B, E]
    bf16*  WxTop  = (bf16*) carve((size_t)EE * H3 * 2);        // [E, 3H]
    bf16*  Wxa    = (bf16*) carve((size_t)MM * H3 * 2);        // [M, 3H]
    bf16*  Whb    = (bf16*) carve((size_t)HH * H3 * 2);        // [H, 3H]
    bf16*  Wmemb  = (bf16*) carve((size_t)MM * HH * 2);        // [M, H]
    bf16*  Wfcb   = (bf16*) carve((size_t)MM * VV * 2);        // [M, V]
    bf16*  memb   = (bf16*) carve((size_t)BB * SS * MM * 2);   // [B*S, M]
    float* mx_pre = (float*)carve((size_t)TT * BB * H3 * 4);   // [T*B, 3H]
    float* keysf  = (float*)carve((size_t)BB * SS * HH * 4);   // [B,S,H]
    float* hbuf   = (float*)carve((size_t)BB * HH * 4);
    bf16*  hb     = (bf16*) carve((size_t)BB * HH * 2);
    float* attnf  = (float*)carve((size_t)BB * MM * 4);
    bf16*  attnb  = (bf16*) carve((size_t)BB * MM * 2);
    float* mxbuf  = (float*)carve((size_t)BB * H3 * 4);
    float* mhbuf  = (float*)carve((size_t)BB * H3 * 4);
    bf16*  ctxb   = (bf16*) carve((size_t)BB * TT * MM * 2);   // [B*T, M]

    auto cvt = [&](const float* s, bf16* d, long n) {
        int blocks = (int)((n + 1023) / 1024);
        if (blocks > 4096) blocks = 4096;
        convert_f32_bf16_kernel<<<blocks, 256, 0, stream>>>(s, d, n);
    };

    // 1) one-shot conversions to bf16
    cvt(W_x,               WxTop, (long)EE * H3);              // rows 0..E-1
    cvt(W_x + (long)EE*H3, Wxa,   (long)MM * H3);              // rows E..E+M-1
    cvt(W_h,               Whb,   (long)HH * H3);
    cvt(W_mem,             Wmemb, (long)MM * HH);
    cvt(W_fc,              Wfcb,  (long)MM * VV);
    cvt(memory,            memb,  (long)BB * SS * MM);

    // 2) embedding gather (bf16 output), [T*B, E]
    {
        long n = (long)TT * BB * EE;
        embed_gather_kernel<<<(int)((n + 255) / 256), 256, 0, stream>>>(tokens, emb, xbf);
    }

    // 3) hoisted input GEMM: mx_pre = x @ W_x[:E] + b_x   [2048 x 3072], K=512
    {
        dim3 grid(H3 / 128, (TT * BB) / 128);
        gemm_bf16_kernel<128, 128, 4, 2><<<grid, 256, 0, stream>>>(
            xbf, WxTop, mx_pre, nullptr, b_x, TT * BB, H3, EE);
    }

    // 4) Luong keys: keys = memory @ W_mem   [2048 x 1024], K=1024
    {
        dim3 grid(HH / 128, (BB * SS) / 128);
        gemm_bf16_kernel<128, 128, 4, 2><<<grid, 256, 0, stream>>>(
            memb, Wmemb, keysf, nullptr, nullptr, BB * SS, HH, MM);
    }

    // 5) zero recurrent state
    init_state_kernel<<<(BB * HH + 255) / 256, 256, 0, stream>>>(hbuf, hb, attnf, attnb);

    // 6) recurrent loop (64 sequential steps)
    for (int t = 0; t < TT; ++t) {
        // mx = attn @ W_x[E:] + mx_pre[t]   [32 x 3072], K=1024
        {
            dim3 grid(H3 / 128, 1);
            gemm_bf16_kernel<32, 128, 2, 4><<<grid, 256, 0, stream>>>(
                attnb, Wxa, mxbuf, mx_pre + (long)t * BB * H3, nullptr,
                BB, H3, MM);
        }
        // mh = h @ W_h + b_h                 [32 x 3072], K=1024
        {
            dim3 grid(H3 / 128, 1);
            gemm_bf16_kernel<32, 128, 2, 4><<<grid, 256, 0, stream>>>(
                hb, Whb, mhbuf, nullptr, b_h, BB, H3, HH);
        }
        // gates + state update
        gru_gate_kernel<<<(BB * HH + 255) / 256, 256, 0, stream>>>(mxbuf, mhbuf, hbuf, hb);
        // attention: score/softmax/context; emits attn + ctxs row t
        attention_step_kernel<<<BB, 256, 0, stream>>>(hbuf, keysf, memory,
                                                      attnf, attnb, ctxb, t);
    }

    // 7) decoder FC: logits = ctxs @ W_fc + b_fc   [2048 x 32000], K=1024
    {
        dim3 grid(VV / 128, (BB * TT) / 128);
        gemm_bf16_kernel<128, 128, 4, 2><<<grid, 256, 0, stream>>>(
            ctxb, Wfcb, logits, nullptr, b_fc, BB * TT, VV, MM);
    }
}
